// SimpleDecoder_10488310136905
// MI455X (gfx1250) — matrix operand order repaired and checked by reading, not compiled, not run on hardware
//
#include <hip/hip_runtime.h>
#include <hip/hip_bf16.h>
#include <math.h>

// ---------------------------------------------------------------------------
// Problem dims
#define D0 128
#define D1 256
#define D2 512
#define D3 256
#define BB 64
#define LL 1024

#define NBLK 64
#define TPB  256
#define NT   (NBLK * TPB)          // 16384 threads, 512 waves

typedef __bf16 bf16_t;
typedef __attribute__((ext_vector_type(16))) __bf16 v16bf;
typedef __attribute__((ext_vector_type(8)))  float  v8f;

// ---------------------------------------------------------------------------
// WMMA helpers (bf16 16x16x32, f32 accumulate)

__device__ __forceinline__ v8f wmma_acc(v16bf a, v16bf b, v8f c) {
    return __builtin_amdgcn_wmma_f32_16x16x32_bf16(
        /*neg_a=*/false, a, /*neg_b=*/false, b,
        /*c_mod=*/(short)0, c, /*reuse_a=*/false, /*reuse_b=*/false);
}

// A-operand 16x32 tile from a row-major matrix (bf16 source).
// Lane l holds row m = l&15; group g = l>>4 selects K-halves:
//   elems 0..7  -> k = c0 + g*8 + e        (contiguous 16B chunk)
//   elems 8..15 -> k = c0 + g*8 + 16 + (e-8) (second contiguous chunk)
__device__ __forceinline__ v16bf load_a_bf16(const bf16_t* base, int ld,
                                             int r0, int c0, int lane) {
    int r = r0 + (lane & 15);
    int c = c0 + ((lane >> 4) << 3);
    const bf16_t* p = base + (size_t)r * ld + c;
    v16bf out;
#pragma unroll
    for (int e = 0; e < 8; ++e)  out[e] = p[e];
#pragma unroll
    for (int e = 8; e < 16; ++e) out[e] = p[e + 8];
    return out;
}

// Same, converting from fp32 source on the fly.
__device__ __forceinline__ v16bf load_a_f32(const float* base, int ld,
                                            int r0, int c0, int lane) {
    int r = r0 + (lane & 15);
    int c = c0 + ((lane >> 4) << 3);
    const float* p = base + (size_t)r * ld + c;
    v16bf out;
#pragma unroll
    for (int e = 0; e < 8; ++e)  out[e] = (bf16_t)p[e];
#pragma unroll
    for (int e = 8; e < 16; ++e) out[e] = (bf16_t)p[e + 8];
    return out;
}

// B operand: weights pre-packed so each lane's 16 bf16 are contiguous (32B).
// Packed index: (((nt*Kt + kt)*32 + lane)*16 + e), lane covers col n0+(l&15),
// k = kt*32 + (l>>4)*8 + (e&7) + (e>>3)*16.
__device__ __forceinline__ v16bf load_b_packed(const bf16_t* Bp, int Kt,
                                               int nt, int kt, int lane) {
    const bf16_t* p = Bp + ((((size_t)nt * Kt + kt) * 32) + lane) * 16;
    return *(const v16bf*)p;
}

// ---------------------------------------------------------------------------
// Grid-wide sense-reversing barrier (persistent kernel phases)
__device__ __forceinline__ void grid_sync(unsigned* bar) {
    __syncthreads();
    if (threadIdx.x == 0) {
        __threadfence();
        volatile unsigned* vcnt = bar;
        volatile unsigned* vgen = bar + 1;
        unsigned g = *vgen;
        if (atomicAdd(bar, 1u) == gridDim.x - 1) {
            *vcnt = 0u;
            __threadfence();
            atomicAdd(bar + 1, 1u);
        } else {
            while (*vgen == g) { __builtin_amdgcn_s_sleep(2); }
        }
        __threadfence();
    }
    __syncthreads();
}

__device__ __forceinline__ float sigmoidf_(float x) {
    return 1.0f / (1.0f + __expf(-x));
}

// ---------------------------------------------------------------------------
// Kernel 1: pack fp32 [K x N] weight -> bf16 WMMA-B tiles (K multiple of 32,
// N multiple of 16).
__global__ void pack_b_kernel(const float* __restrict__ src,
                              bf16_t* __restrict__ dst, int K, int N) {
    int idx = blockIdx.x * blockDim.x + threadIdx.x;
    if (idx >= K * N) return;
    int e    = idx & 15;
    int lane = (idx >> 4) & 31;
    int tile = idx >> 9;
    int Kt   = K >> 5;
    int nt   = tile / Kt;
    int kt   = tile - nt * Kt;
    int n = nt * 16 + (lane & 15);
    int k = kt * 32 + ((lane >> 4) << 3) + (e & 7) + ((e >> 3) << 4);
    dst[idx] = (bf16_t)src[(size_t)k * N + n];
}

// ---------------------------------------------------------------------------
// Kernel 2: K/V projection. k = hid_r@Wk + bk, v = hid_r@Wv + bv, bf16 out
// (kept L2-resident: 64 MB of the 192 MB L2). One wave per 16x16 tile; the
// A tiles are shared between the K and V WMMA chains.
__global__ void kv_gemm_kernel(const float* __restrict__ hid_r,
                               const bf16_t* __restrict__ Wkp,
                               const bf16_t* __restrict__ Wvp,
                               const float* __restrict__ bk,
                               const float* __restrict__ bv,
                               bf16_t* __restrict__ kout,
                               bf16_t* __restrict__ vout) {
    int gtid = blockIdx.x * blockDim.x + threadIdx.x;
    int wid  = gtid >> 5;
    int lane = gtid & 31;
    int mt = wid >> 4;          // 4096 row tiles of (B*L)=65536
    int nt = wid & 15;          // 16 col tiles of D3=256
    int m0 = mt * 16;
    v8f ka = {};
    v8f va = {};
    const int Kt = D3 / 32;     // 8
#pragma unroll
    for (int kt = 0; kt < Kt; ++kt) {
        v16bf a = load_a_f32(hid_r, D3, m0, kt * 32, lane);
        ka = wmma_acc(a, load_b_packed(Wkp, Kt, nt, kt, lane), ka);
        va = wmma_acc(a, load_b_packed(Wvp, Kt, nt, kt, lane), va);
    }
    int n  = nt * 16 + (lane & 15);
    int mg = (lane >> 4) << 3;
    float kb = bk[n], vb = bv[n];
#pragma unroll
    for (int r = 0; r < 8; ++r) {
        size_t m = (size_t)(m0 + mg + r);
        kout[m * D3 + n] = (bf16_t)(ka[r] + kb);
        vout[m * D3 + n] = (bf16_t)(va[r] + vb);
    }
}

// ---------------------------------------------------------------------------
// Kernel 3: initialize recurrent state + barrier counters.
__global__ void init_state_kernel(const float* __restrict__ init_in,
                                  const float* __restrict__ h0,
                                  const float* __restrict__ c0,
                                  bf16_t* __restrict__ x_bf,
                                  bf16_t* __restrict__ h_bf,
                                  float* __restrict__ c,
                                  unsigned* __restrict__ bar) {
    int i = blockIdx.x * blockDim.x + threadIdx.x;
    int nt = gridDim.x * blockDim.x;
    for (int j = i; j < BB * D0; j += nt) x_bf[j] = (bf16_t)init_in[j];
    for (int j = i; j < BB * D2; j += nt) {
        h_bf[j] = (bf16_t)h0[j];
        c[j]    = c0[j];
    }
    if (i < 8) bar[i] = 0u;
}

// ---------------------------------------------------------------------------
// Kernel 4: persistent decoder. 64 blocks x 256 threads (512 wave32 waves),
// loops the 1024-step scan on chip; phases separated by grid barrier.
struct DecArgs {
    const float  *in_mask, *b1, *bih, *bhh, *b2, *bq, *b3;
    const bf16_t *W1p, *Wihp, *Whhp, *W2p, *Wqp, *W3p;
    const bf16_t *kmat, *vmat;
    bf16_t *x_bf, *h_bf, *a_bf, *dx_bf;
    float  *c, *gates, *q, *scores, *attn, *ctx;
    float  *out_y, *out_attn;
    unsigned *bar;
};

__global__ void __launch_bounds__(TPB, 1) decoder_kernel(DecArgs A) {
    extern __shared__ char smem[];   // 96 KB: a (32KB) + h (64KB) staging
    const int tid  = threadIdx.x;
    const int gtid = blockIdx.x * TPB + tid;
    const int wid  = gtid >> 5;
    const int lane = gtid & 31;

    for (int t = 0; t < LL; ++t) {
        // ---- Phase 1: a = relu(x @ W1 + b1)  [64x128]@[128x256]
        if (wid < (BB / 16) * (D1 / 16)) {
            int mt = wid >> 4, nt = wid & 15;
            v8f acc = {};
            const int Kt = D0 / 32;   // 4
#pragma unroll
            for (int kt = 0; kt < Kt; ++kt)
                acc = wmma_acc(load_a_bf16(A.x_bf, D0, mt * 16, kt * 32, lane),
                               load_b_packed(A.W1p, Kt, nt, kt, lane), acc);
            int n = nt * 16 + (lane & 15), mg = (lane >> 4) << 3;
            float bias = A.b1[n];
#pragma unroll
            for (int r = 0; r < 8; ++r) {
                float y = acc[r] + bias;
                y = y > 0.f ? y : 0.f;
                A.a_bf[(mt * 16 + mg + r) * D1 + n] = (bf16_t)y;
            }
        }
        grid_sync(A.bar);

        // ---- Phase 2: gates = a@Wih + h@Whh + bih + bhh  -> [64x2048]
        {
            bf16_t* a_sh = (bf16_t*)smem;                       // 64x256
            bf16_t* h_sh = (bf16_t*)(smem + BB * D1 * 2);       // 64x512
            const uint4* srcA = (const uint4*)A.a_bf;
            uint4* dstA = (uint4*)a_sh;
            for (int j = tid; j < (BB * D1 * 2) / 16; j += TPB) dstA[j] = srcA[j];
            const uint4* srcH = (const uint4*)A.h_bf;
            uint4* dstH = (uint4*)h_sh;
            for (int j = tid; j < (BB * D2 * 2) / 16; j += TPB) dstH[j] = srcH[j];
            __syncthreads();

            int mt = wid >> 7, nt = wid & 127;                  // 512 tiles
            v8f acc = {};
            const int Kta = D1 / 32;   // 8
            const int Kth = D2 / 32;   // 16
#pragma unroll
            for (int kt = 0; kt < Kta; ++kt)
                acc = wmma_acc(load_a_bf16(a_sh, D1, mt * 16, kt * 32, lane),
                               load_b_packed(A.Wihp, Kta, nt, kt, lane), acc);
#pragma unroll
            for (int kt = 0; kt < Kth; ++kt)
                acc = wmma_acc(load_a_bf16(h_sh, D2, mt * 16, kt * 32, lane),
                               load_b_packed(A.Whhp, Kth, nt, kt, lane), acc);
            int n = nt * 16 + (lane & 15), mg = (lane >> 4) << 3;
            float bias = A.bih[n] + A.bhh[n];
#pragma unroll
            for (int r = 0; r < 8; ++r)
                A.gates[(mt * 16 + mg + r) * (4 * D2) + n] = acc[r] + bias;
        }
        grid_sync(A.bar);

        // ---- Phase 3: LSTM pointwise  (c, h update)
        for (int i = gtid; i < BB * D2; i += NT) {
            int b = i >> 9, d = i & (D2 - 1);
            const float* g = A.gates + b * (4 * D2);
            float ig = g[d], fg = g[D2 + d], gg = g[2 * D2 + d], og = g[3 * D2 + d];
            float cn = sigmoidf_(fg) * A.c[i] + sigmoidf_(ig) * tanhf(gg);
            float hn = sigmoidf_(og) * tanhf(cn);
            A.c[i] = cn;
            A.h_bf[i] = (bf16_t)hn;
        }
        grid_sync(A.bar);

        // ---- Phase 4: dx = relu(h @ W2 + b2)  [64x512]@[512x256]
        if (wid < (BB / 16) * (D3 / 16)) {
            int mt = wid >> 4, nt = wid & 15;
            v8f acc = {};
            const int Kt = D2 / 32;   // 16
#pragma unroll
            for (int kt = 0; kt < Kt; ++kt)
                acc = wmma_acc(load_a_bf16(A.h_bf, D2, mt * 16, kt * 32, lane),
                               load_b_packed(A.W2p, Kt, nt, kt, lane), acc);
            int n = nt * 16 + (lane & 15), mg = (lane >> 4) << 3;
            float bias = A.b2[n];
#pragma unroll
            for (int r = 0; r < 8; ++r) {
                float y = acc[r] + bias;
                y = y > 0.f ? y : 0.f;
                A.dx_bf[(mt * 16 + mg + r) * D3 + n] = (bf16_t)y;
            }
        }
        grid_sync(A.bar);

        // ---- Phase 5: q = dx @ Wq + bq  (fp32 out)
        if (wid < (BB / 16) * (D3 / 16)) {
            int mt = wid >> 4, nt = wid & 15;
            v8f acc = {};
            const int Kt = D3 / 32;   // 8
#pragma unroll
            for (int kt = 0; kt < Kt; ++kt)
                acc = wmma_acc(load_a_bf16(A.dx_bf, D3, mt * 16, kt * 32, lane),
                               load_b_packed(A.Wqp, Kt, nt, kt, lane), acc);
            int n = nt * 16 + (lane & 15), mg = (lane >> 4) << 3;
            float bias = A.bq[n];
#pragma unroll
            for (int r = 0; r < 8; ++r)
                A.q[(mt * 16 + mg + r) * D3 + n] = acc[r] + bias;
        }
        grid_sync(A.bar);

        // ---- Phase 6a: scores[b,l] = scale * q[b]·k[b,l] + mask_bias
        {
            const float scale = 0.0625f;   // 1/sqrt(256)
            for (int i = gtid; i < BB * LL; i += NT) {
                int b = i >> 10, l = i & (LL - 1);
                const bf16_t* kp = A.kmat + ((size_t)(b << 10) + l) * D3;
                const float*  qp = A.q + b * D3;
                float s = 0.f;
#pragma unroll 8
                for (int d = 0; d < D3; ++d) s += (float)kp[d] * qp[d];
                float mb = (A.in_mask[(b << 10) + l] == 0.f) ? -1.0e9f : 0.f;
                A.scores[i] = s * scale + mb;
            }
        }
        grid_sync(A.bar);

        // ---- Phase 6b: softmax over l (block b owns row b)
        {
            float* red = (float*)smem;
            int b = blockIdx.x;
            const float* sr = A.scores + (b << 10);
            float sv[4];
            float mx = -3.4e38f;
#pragma unroll
            for (int j = 0; j < 4; ++j) {
                sv[j] = sr[tid + j * TPB];
                mx = fmaxf(mx, sv[j]);
            }
            red[tid] = mx; __syncthreads();
            for (int s = TPB / 2; s > 0; s >>= 1) {
                if (tid < s) red[tid] = fmaxf(red[tid], red[tid + s]);
                __syncthreads();
            }
            mx = red[0]; __syncthreads();
            float sum = 0.f;
#pragma unroll
            for (int j = 0; j < 4; ++j) { sv[j] = __expf(sv[j] - mx); sum += sv[j]; }
            red[tid] = sum; __syncthreads();
            for (int s = TPB / 2; s > 0; s >>= 1) {
                if (tid < s) red[tid] += red[tid + s];
                __syncthreads();
            }
            float inv = 1.f / red[0]; __syncthreads();
            float* oa = A.out_attn + ((size_t)(b << 10) + t) * LL;
#pragma unroll
            for (int j = 0; j < 4; ++j) {
                float av = sv[j] * inv;
                A.attn[(b << 10) + tid + j * TPB] = av;
                oa[tid + j * TPB] = av;
            }
        }
        grid_sync(A.bar);

        // ---- Phase 7: ctx[b,d] = sum_l attn[b,l] * v[b,l,d]
        for (int i = gtid; i < BB * D3; i += NT) {
            int b = i >> 8, d = i & (D3 - 1);
            const bf16_t* vp = A.vmat + ((size_t)b << 10) * D3 + d;
            const float*  ap = A.attn + (b << 10);
            float s = 0.f;
#pragma unroll 4
            for (int l = 0; l < LL; ++l) s += ap[l] * (float)vp[(size_t)l * D3];
            A.ctx[i] = s;
        }
        grid_sync(A.bar);

        // ---- Phase 8: y = relu(ctx @ W3 + b3); write output + next x
        if (wid < (BB / 16) * (D0 / 16)) {   // 32 tiles
            int mt = wid >> 3, nt = wid & 7;
            v8f acc = {};
            const int Kt = D3 / 32;   // 8
#pragma unroll
            for (int kt = 0; kt < Kt; ++kt)
                acc = wmma_acc(load_a_f32(A.ctx, D3, mt * 16, kt * 32, lane),
                               load_b_packed(A.W3p, Kt, nt, kt, lane), acc);
            int n = nt * 16 + (lane & 15), mg = (lane >> 4) << 3;
            float bias = A.b3[n];
#pragma unroll
            for (int r = 0; r < 8; ++r) {
                int m = mt * 16 + mg + r;
                float y = acc[r] + bias;
                y = y > 0.f ? y : 0.f;
                A.x_bf[m * D0 + n] = (bf16_t)y;
                A.out_y[((size_t)m * LL + t) * D0 + n] = y * A.in_mask[(m << 10) + t];
            }
        }
        grid_sync(A.bar);
    }
}

// ---------------------------------------------------------------------------
extern "C" void kernel_launch(void* const* d_in, const int* in_sizes, int n_in,
                              void* d_out, int out_size, void* d_ws, size_t ws_size,
                              hipStream_t stream) {
    (void)in_sizes; (void)n_in; (void)out_size; (void)ws_size;
    const float* hid_r   = (const float*)d_in[0];
    const float* in_mask = (const float*)d_in[1];
    const float* init_in = (const float*)d_in[2];
    const float* h0      = (const float*)d_in[3];
    const float* c0      = (const float*)d_in[4];
    const float* W1      = (const float*)d_in[5];
    const float* b1      = (const float*)d_in[6];
    const float* Wih     = (const float*)d_in[7];
    const float* Whh     = (const float*)d_in[8];
    const float* bih     = (const float*)d_in[9];
    const float* bhh     = (const float*)d_in[10];
    const float* W2      = (const float*)d_in[11];
    const float* b2      = (const float*)d_in[12];
    const float* Wq      = (const float*)d_in[13];
    const float* bq      = (const float*)d_in[14];
    const float* Wk      = (const float*)d_in[15];
    const float* bk      = (const float*)d_in[16];
    const float* Wv      = (const float*)d_in[17];
    const float* bv      = (const float*)d_in[18];
    const float* W3      = (const float*)d_in[19];
    const float* b3      = (const float*)d_in[20];

    char* ws = (char*)d_ws;
    size_t off = 0;
    auto alloc = [&](size_t bytes) -> char* {
        char* p = ws + off;
        off = (off + bytes + 255) & ~(size_t)255;
        return p;
    };

    bf16_t* W1p  = (bf16_t*)alloc((size_t)D0 * D1 * 2);
    bf16_t* Wihp = (bf16_t*)alloc((size_t)D1 * 4 * D2 * 2);
    bf16_t* Whhp = (bf16_t*)alloc((size_t)D2 * 4 * D2 * 2);
    bf16_t* W2p  = (bf16_t*)alloc((size_t)D2 * D3 * 2);
    bf16_t* Wqp  = (bf16_t*)alloc((size_t)D3 * D3 * 2);
    bf16_t* Wkp  = (bf16_t*)alloc((size_t)D3 * D3 * 2);
    bf16_t* Wvp  = (bf16_t*)alloc((size_t)D3 * D3 * 2);
    bf16_t* W3p  = (bf16_t*)alloc((size_t)D3 * D0 * 2);
    bf16_t* kmat = (bf16_t*)alloc((size_t)BB * LL * D3 * 2);
    bf16_t* vmat = (bf16_t*)alloc((size_t)BB * LL * D3 * 2);
    bf16_t* x_bf = (bf16_t*)alloc((size_t)BB * D0 * 2);
    bf16_t* h_bf = (bf16_t*)alloc((size_t)BB * D2 * 2);
    bf16_t* a_bf = (bf16_t*)alloc((size_t)BB * D1 * 2);
    bf16_t* dx_bf= (bf16_t*)alloc((size_t)BB * D3 * 2);
    float*  c    = (float*)alloc((size_t)BB * D2 * 4);
    float*  gates= (float*)alloc((size_t)BB * 4 * D2 * 4);
    float*  q    = (float*)alloc((size_t)BB * D3 * 4);
    float*  scr  = (float*)alloc((size_t)BB * LL * 4);
    float*  attn = (float*)alloc((size_t)BB * LL * 4);
    float*  ctx  = (float*)alloc((size_t)BB * D3 * 4);
    unsigned* bar= (unsigned*)alloc(256);

    // 1) pack all weights into WMMA-B bf16 layout
    auto pack = [&](const float* src, bf16_t* dst, int K, int N) {
        int total = K * N;
        pack_b_kernel<<<(total + 255) / 256, 256, 0, stream>>>(src, dst, K, N);
    };
    pack(W1,  W1p,  D0, D1);
    pack(Wih, Wihp, D1, 4 * D2);
    pack(Whh, Whhp, D2, 4 * D2);
    pack(W2,  W2p,  D2, D3);
    pack(Wq,  Wqp,  D3, D3);
    pack(Wk,  Wkp,  D3, D3);
    pack(Wv,  Wvp,  D3, D3);
    pack(W3,  W3p,  D3, D0);

    // 2) K/V projection (L2-resident bf16), one wave per 16x16 tile
    {
        int tiles = (BB * LL / 16) * (D3 / 16);   // 65536 waves
        int blocks = (tiles * 32) / 256;          // 8192
        kv_gemm_kernel<<<blocks, 256, 0, stream>>>(hid_r, Wkp, Wvp, bk, bv,
                                                   kmat, vmat);
    }

    // 3) init recurrent state + barrier
    init_state_kernel<<<64, 256, 0, stream>>>(init_in, h0, c0, x_bf, h_bf, c, bar);

    // 4) persistent scan kernel
    DecArgs A;
    A.in_mask = in_mask; A.b1 = b1; A.bih = bih; A.bhh = bhh;
    A.b2 = b2; A.bq = bq; A.b3 = b3;
    A.W1p = W1p; A.Wihp = Wihp; A.Whhp = Whhp; A.W2p = W2p;
    A.Wqp = Wqp; A.W3p = W3p;
    A.kmat = kmat; A.vmat = vmat;
    A.x_bf = x_bf; A.h_bf = h_bf; A.a_bf = a_bf; A.dx_bf = dx_bf;
    A.c = c; A.gates = gates; A.q = q; A.scores = scr; A.attn = attn; A.ctx = ctx;
    A.out_y = (float*)d_out;
    A.out_attn = (float*)d_out + (size_t)BB * LL * D0;
    A.bar = bar;

    size_t shmem = (size_t)BB * D1 * 2 + (size_t)BB * D2 * 2;   // 96 KB
    decoder_kernel<<<NBLK, TPB, shmem, stream>>>(A);
}
